// CompleteDTBackbone_44770739093812
// MI455X (gfx1250) — matrix operand order repaired and checked by reading, not compiled, not run on hardware
//
#include <hip/hip_runtime.h>
#include <math.h>

#define BB 8
#define NN 8192
#define C0 64
#define C1 128
#define KNN 32
#define M1 2048
#define M2 512
#define CIN 134
#define CINP 160   // K padded to multiple of 32 for WMMA

typedef unsigned short u16;
typedef unsigned int   u32;
typedef unsigned long long u64;

typedef __attribute__((ext_vector_type(16))) __bf16 v16bf;
typedef __attribute__((ext_vector_type(16))) u16    v16u;
typedef __attribute__((ext_vector_type(8)))  float  v8f;
typedef __attribute__((ext_vector_type(4)))  u32    u32x4;
typedef __attribute__((ext_vector_type(8)))  int    i32x8;
typedef __attribute__((ext_vector_type(4)))  int    i32x4;

// ---- output layout (float elements), tuple order of the reference ----
#define O_PL0  0          // [8,3,8192]
#define O_NEW1 196608     // [8,128,2048]
#define O_PL2  2293760    // [8,3,512]
#define O_FL0  2306048    // [8,64,8192]
#define O_FL1  6500352    // [8,64,2048]
#define O_FL2  7548928    // [8,64,512]
#define O_GIDX 7811072    // [8,2048,32] (stored as float)

__device__ __forceinline__ float gelu_f(float x) {
  return 0.5f * x * (1.0f + erff(x * 0.70710678118654752440f));
}
__device__ __forceinline__ u16 f2bf(float f) {
  u32 u = __float_as_uint(f);
  u += 0x7FFFu + ((u >> 16) & 1u);   // round-to-nearest-even
  return (u16)(u >> 16);
}
__device__ __forceinline__ float bf2f(u16 h) {
  return __uint_as_float(((u32)h) << 16);
}

// ---------------------------------------------------------------------
// TDM: async DMA of a contiguous run of u16 elements (global -> LDS).
// D# per CDNA5 ISA ch.8: group0 = {count=1, lds_addr, global_addr, type=2},
// group1 = {data_size=2B, tensor_dim0 = n, tensor_dim1 = 1, tile_dim0 = n,
//           tile_dim1 = 1, stride0 = n}; groups 2/3 zero (2-D tensor).
// ---------------------------------------------------------------------
__device__ __forceinline__ void tdm_load_u16(u32 lds_off, const void* gptr, u32 n) {
  u64 ga = (u64)(size_t)gptr;
  u32x4 g0;
  g0[0] = 1u;                                   // count = 1, user descriptor
  g0[1] = lds_off;                              // lds_addr (bytes)
  g0[2] = (u32)ga;                              // global_addr[31:0]
  g0[3] = (u32)(ga >> 32) | 0x80000000u;        // global_addr[56:32] | type=2<<30
  i32x8 g1;
  g1[0] = (int)(1u << 16);                      // data_size = 1 (2 bytes)
  g1[1] = (int)((n & 0xFFFFu) << 16);           // tensor_dim0[15:0] in bits 63:48
  g1[2] = (int)((n >> 16) | (1u << 16));        // tensor_dim0[31:16] | tensor_dim1=1
  g1[3] = (int)((n & 0xFFFFu) << 16);           // tile_dim0 in bits 127:112
  g1[4] = 1;                                    // tile_dim1 = 1
  g1[5] = (int)n;                               // tensor_dim0_stride[31:0]
  g1[6] = 0;
  g1[7] = 0;
  i32x4 z4 = {0, 0, 0, 0};
#if defined(__clang_major__) && (__clang_major__ >= 23)
  i32x8 z8 = {0, 0, 0, 0, 0, 0, 0, 0};
  __builtin_amdgcn_tensor_load_to_lds(g0, g1, z4, z4, z8, 0);
#else
  __builtin_amdgcn_tensor_load_to_lds(g0, g1, z4, z4, 0);
#endif
}

// ---------------------------------------------------------------------
// 0) Fold BN scales into weights; bf16-convert + K-pad the grouped-conv weights
// ---------------------------------------------------------------------
__global__ void k_prep(const float* W1, const float* s1, const float* W2, const float* s2,
                       const float* Wp, const float* sp, const float* Wg, const float* sg,
                       const float* Wr1, const float* sr1, const float* Wr2, const float* sr2,
                       float* W1f, float* W2f, float* Wpf, u16* Wgb, u16* Wr1b, u16* Wr2b) {
  const int total = 192 + 4096 + 4096 + 128 * CINP + 16384 + 16384;
  for (int j = blockIdx.x * blockDim.x + threadIdx.x; j < total; j += gridDim.x * blockDim.x) {
    int t = j;
    if (t < 192) { int o = t / 3; W1f[t] = s1[o] * W1[t]; continue; } t -= 192;
    if (t < 4096) { int o = t >> 6; W2f[t] = s2[o] * W2[t]; continue; } t -= 4096;
    if (t < 4096) { int o = t >> 6; Wpf[t] = sp[o] * Wp[t]; continue; } t -= 4096;
    if (t < 128 * CINP) {
      int o = t / CINP, k = t % CINP;
      float v = (k < CIN) ? sg[o] * Wg[o * CIN + k] : 0.0f;
      Wgb[t] = f2bf(v); continue;
    } t -= 128 * CINP;
    if (t < 16384) { int o = t >> 7; Wr1b[t] = f2bf(sr1[o] * Wr1[t]); continue; } t -= 16384;
    { int o = t >> 7; Wr2b[t] = f2bf(sr2[o] * Wr2[t]); }
  }
}

// ---------------------------------------------------------------------
// 1) pl[0] = points (copy)
// ---------------------------------------------------------------------
__global__ void k_copy(const float* pts, float* out) {
  for (int i = blockIdx.x * blockDim.x + threadIdx.x; i < BB * 3 * NN; i += gridDim.x * blockDim.x)
    out[O_PL0 + i] = pts[i];
}

// ---------------------------------------------------------------------
// 2) feats = gelu(conv2(gelu(conv1(points)))) + gelu(conv1(points))  -> fl[0]
// ---------------------------------------------------------------------
__global__ void __launch_bounds__(256) k_feats(const float* pts, const float* W1f, const float* b1,
                                               const float* W2f, const float* b2, float* out) {
  __shared__ float sW1[192], sB1[64], sW2[4096], sB2[64];
  for (int i = threadIdx.x; i < 192; i += blockDim.x) sW1[i] = W1f[i];
  for (int i = threadIdx.x; i < 64; i += blockDim.x) { sB1[i] = b1[i]; sB2[i] = b2[i]; }
  for (int i = threadIdx.x; i < 4096; i += blockDim.x) sW2[i] = W2f[i];
  __syncthreads();
  int p = blockIdx.x * blockDim.x + threadIdx.x;     // 0..65535
  int b = p >> 13, n = p & (NN - 1);
  const float* pb = pts + b * 3 * NN;
  float x = pb[n], y = pb[NN + n], z = pb[2 * NN + n];
  float h0[64];
#pragma unroll
  for (int o = 0; o < 64; o++)
    h0[o] = gelu_f(sW1[o * 3] * x + sW1[o * 3 + 1] * y + sW1[o * 3 + 2] * z + sB1[o]);
  float* ob = out + O_FL0 + (size_t)b * 64 * NN + n;
#pragma unroll 4
  for (int o = 0; o < 64; o++) {
    float acc = sB2[o];
#pragma unroll
    for (int k = 0; k < 64; k++) acc += sW2[o * 64 + k] * h0[k];
    ob[(size_t)o * NN] = gelu_f(acc) + h0[o];
  }
}

// ---------------------------------------------------------------------
// 3) Furthest-point sampling: one workgroup per batch, xyz in LDS,
//    per-thread min-dists in registers, argmax via packed u64 keys.
//    (the 512-sample FPS is a prefix of the 2048-sample FPS)
// ---------------------------------------------------------------------
__global__ void __launch_bounds__(1024) k_fps(const float* pts, int* fpsidx) {
  __shared__ float lx[NN], ly[NN], lz[NN];
  __shared__ u64 red[32];
  __shared__ int sCur;
  int b = blockIdx.x, tid = threadIdx.x;
  const float* pb = pts + b * 3 * NN;
  for (int i = tid; i < NN; i += 1024) { lx[i] = pb[i]; ly[i] = pb[NN + i]; lz[i] = pb[2 * NN + i]; }
  float dloc[8];
#pragma unroll
  for (int j = 0; j < 8; j++) dloc[j] = 1e10f;
  if (tid == 0) { fpsidx[b * M1] = 0; sCur = 0; }
  __syncthreads();
  int cur = 0;
  for (int s = 1; s < M1; s++) {
    float cx = lx[cur], cy = ly[cur], cz = lz[cur];
    u64 best = 0;
#pragma unroll
    for (int j = 0; j < 8; j++) {
      int i = tid + 1024 * j;
      float dx = lx[i] - cx, dy = ly[i] - cy, dz = lz[i] - cz;
      float d = dx * dx + dy * dy + dz * dz;
      float m = fminf(dloc[j], d); dloc[j] = m;
      u64 key = ((u64)__float_as_uint(m) << 32) | (u32)(~(u32)i); // argmax, ties -> lowest idx
      best = best > key ? best : key;
    }
    for (int o = 16; o > 0; o >>= 1) { u64 v = __shfl_xor(best, o, 32); best = best > v ? best : v; }
    if ((tid & 31) == 0) red[tid >> 5] = best;
    __syncthreads();
    if (tid < 32) {
      u64 v = red[tid];
      for (int o = 16; o > 0; o >>= 1) { u64 w = __shfl_xor(v, o, 32); v = v > w ? v : w; }
      if (tid == 0) { int nx = (int)(~(u32)(v & 0xFFFFFFFFull)); sCur = nx; fpsidx[b * M1 + s] = nx; }
    }
    __syncthreads();
    cur = sCur;
  }
}

// ---------------------------------------------------------------------
// 4) Gathers: centers (ws) + pl[2], fl[1], fl[2]
// ---------------------------------------------------------------------
__global__ void k_gather(const float* pts, const int* fpsidx, float* out, float* ctr) {
  const int totA = BB * 3 * M1, totB = BB * 64 * M1;
  for (int j = blockIdx.x * blockDim.x + threadIdx.x; j < totA + totB; j += gridDim.x * blockDim.x) {
    if (j < totA) {
      int b = j / (3 * M1), r = j % (3 * M1), c = r / M1, m = r % M1;
      int gi = fpsidx[b * M1 + m];
      float v = pts[(b * 3 + c) * NN + gi];
      ctr[(b * 3 + c) * M1 + m] = v;
      if (m < M2) out[O_PL2 + (b * 3 + c) * M2 + m] = v;
    } else {
      int t = j - totA;
      int b = t / (64 * M1), ch = (t / M1) & 63, m = t % M1;
      int gi = fpsidx[b * M1 + m];
      float v = out[O_FL0 + (size_t)(b * 64 + ch) * NN + gi];
      out[O_FL1 + (size_t)(b * 64 + ch) * M1 + m] = v;
      if (m < M2) out[O_FL2 + (b * 64 + ch) * M2 + m] = v;
    }
  }
}

// ---------------------------------------------------------------------
// 5) fp = gelu(Wp * fl[1] + bp)   (64x64 per-center)
// ---------------------------------------------------------------------
__global__ void __launch_bounds__(256) k_fp(const float* dout, const float* Wpf, const float* bp, float* fp) {
  __shared__ float sW[4096], sB[64];
  for (int i = threadIdx.x; i < 4096; i += blockDim.x) sW[i] = Wpf[i];
  for (int i = threadIdx.x; i < 64; i += blockDim.x) sB[i] = bp[i];
  __syncthreads();
  int t = blockIdx.x * blockDim.x + threadIdx.x;     // 0..16383
  int b = t / M1, m = t % M1;
  const float* ib = dout + O_FL1 + (size_t)b * 64 * M1 + m;
  float h[64];
#pragma unroll
  for (int k = 0; k < 64; k++) h[k] = ib[(size_t)k * M1];
#pragma unroll 4
  for (int o = 0; o < 64; o++) {
    float a = sB[o];
#pragma unroll
    for (int k = 0; k < 64; k++) a += sW[o * 64 + k] * h[k];
    fp[(size_t)(b * 64 + o) * M1 + m] = gelu_f(a);
  }
}

// ---------------------------------------------------------------------
// 6) kNN (k=32): one block per center; per-lane candidate keys in registers,
//    32 rounds of packed-u64 min extraction (ascending d2, low-index ties)
// ---------------------------------------------------------------------
__global__ void __launch_bounds__(256) k_knn(const float* pts, const float* ctr, float* out, int* gidx) {
  __shared__ u64 sd[256];
  int blk = blockIdx.x, b = blk / M1, m = blk % M1, tid = threadIdx.x;
  float cx = ctr[(b * 3 + 0) * M1 + m];
  float cy = ctr[(b * 3 + 1) * M1 + m];
  float cz = ctr[(b * 3 + 2) * M1 + m];
  const float* pb = pts + b * 3 * NN;
  u64 key[32];
#pragma unroll
  for (int t = 0; t < 32; t++) {
    int i = tid + 256 * t;
    float dx = pb[i] - cx, dy = pb[NN + i] - cy, dz = pb[2 * NN + i] - cz;
    float d = dx * dx + dy * dy + dz * dz;
    key[t] = ((u64)__float_as_uint(d) << 32) | (u32)i;
  }
  for (int r = 0; r < KNN; r++) {
    u64 lm = key[0];
#pragma unroll
    for (int t = 1; t < 32; t++) lm = key[t] < lm ? key[t] : lm;
    sd[tid] = lm;
    __syncthreads();
    for (int off = 128; off > 0; off >>= 1) {
      if (tid < off) { u64 o2 = sd[tid + off]; if (o2 < sd[tid]) sd[tid] = o2; }
      __syncthreads();
    }
    u64 win = sd[0];
    if (tid == 0) {
      int gi = (int)(u32)(win & 0xFFFFFFFFull);
      gidx[(b * M1 + m) * KNN + r] = gi;
      out[O_GIDX + (size_t)(b * M1 + m) * KNN + r] = (float)gi;
    }
#pragma unroll
    for (int t = 0; t < 32; t++) if (key[t] == win) key[t] = 0xFFFFFFFFFFFFFFFFull;
    __syncthreads();
  }
}

// ---------------------------------------------------------------------
// 7) Fused grouped-conv chain via v_wmma_f32_16x16x32_bf16.
//    Weights DMA'd into LDS by the Tensor Data Mover (overlapped with the
//    X-tile gather), then 3 chained GEMMs fully in LDS, then max over K.
//    ISA 7.12.2 fragment layouts:
//      A (16x32, MxK): M = lane&15; elem e -> K = (e>=8?16:0) + 8*(lane>=16) + (e&7)
//      B (32x16, KxN): N = lane&15; elem e -> K = (e>=8?16:0) + 8*(lane>=16) + (e&7)
//      D (16x16 f32):  N = lane&15; reg r -> M = r + 8*(lane>=16)
// ---------------------------------------------------------------------
template <int NKT, int NCT, int KPA, int SB, int ACT, int RES>
__device__ __forceinline__ void gemm_block(const u16* sA, const u16* sB, u16* sD,
                                           const u16* sRes, const float* bias, int wv, int ln) {
  int mm = ln & 15, hi = ln >> 4;
  v16u afr[NKT];
#pragma unroll
  for (int kt = 0; kt < NKT; kt++) {
#pragma unroll
    for (int e = 0; e < 16; e++) {
      int k = kt * 32 + ((e >= 8) ? 16 : 0) + hi * 8 + (e & 7);
      afr[kt][e] = sA[(wv * 16 + mm) * KPA + k];
    }
  }
#pragma unroll
  for (int ct = 0; ct < NCT; ct++) {
    v8f acc = {0.f, 0.f, 0.f, 0.f, 0.f, 0.f, 0.f, 0.f};
    int col = ct * 16 + mm;
#pragma unroll
    for (int kt = 0; kt < NKT; kt++) {
      v16u bfr;
#pragma unroll
      for (int e = 0; e < 16; e++) bfr[e] = sB[col * SB + kt * 32 + ((e >= 8) ? 16 : 0) + hi * 8 + (e & 7)];
      acc = __builtin_amdgcn_wmma_f32_16x16x32_bf16(
          false, __builtin_bit_cast(v16bf, afr[kt]),
          false, __builtin_bit_cast(v16bf, bfr),
          (short)0, acc, false, false);
    }
#pragma unroll
    for (int r = 0; r < 8; r++) {
      int o = wv * 16 + r + hi * 8;
      float v = acc[r] + bias[o];
      if (ACT) v = gelu_f(v);
      if (RES) v += bf2f(sRes[col * 128 + o]);
      sD[col * 128 + o] = f2bf(v);
    }
  }
}

__global__ void __launch_bounds__(256) k_groupconv(
    const float* dout_c, const float* pts, const float* ctr, const float* fp, const int* gidx,
    const u16* Wgb, const float* bg, const u16* Wr1b, const float* br1,
    const u16* Wr2b, const float* br2, float* out) {
  __shared__ u16 sWg[128 * CINP];    // 40960 B
  __shared__ u16 sWr1[128 * 128];    // 32768 B
  __shared__ u16 sWr2[128 * 128];    // 32768 B
  __shared__ u16 sX[64 * CINP];      // 20480 B  (input X, reused as Y2 w/ stride 128)
  __shared__ u16 sY1[64 * 128];      // 16384 B
  int b = blockIdx.x >> 10, m0 = (blockIdx.x & 1023) * 2, tid = threadIdx.x;

  // Wave 0 issues the three weight DMAs (TDM); tracked by TENSORcnt.
  if (tid < 32) {
    tdm_load_u16((u32)(size_t)(void*)sWg,  Wgb,  128 * CINP);
    tdm_load_u16((u32)(size_t)(void*)sWr1, Wr1b, 128 * 128);
    tdm_load_u16((u32)(size_t)(void*)sWr2, Wr2b, 128 * 128);
  }

  // Meanwhile: build X with all threads (gather-heavy, overlaps the DMA).
  // columns = 2 centers x 32 neighbors; channels = [gf(64) | fp(64) | gp-ctr(3) | ctr(3) | pad]
  for (int t = tid; t < 64 * CINP; t += 256) {
    int col = t / CINP, ch = t % CINP;
    int c = m0 + (col >> 5), kk = col & 31;
    int gi = gidx[(b * M1 + c) * KNN + kk];
    float v = 0.f;
    if (ch < 64)       v = dout_c[O_FL0 + (size_t)(b * 64 + ch) * NN + gi];
    else if (ch < 128) v = fp[(size_t)(b * 64 + (ch - 64)) * M1 + c];
    else if (ch < 131) { int d = ch - 128; v = pts[(b * 3 + d) * NN + gi] - ctr[(b * 3 + d) * M1 + c]; }
    else if (ch < 134) { int d = ch - 131; v = ctr[(b * 3 + d) * M1 + c]; }
    sX[col * CINP + ch] = f2bf(v);
  }

  // Wave 0 drains its TDM ops, then the whole workgroup synchronizes.
  if (tid < 32) __builtin_amdgcn_s_wait_tensorcnt(0);
  __syncthreads();

  int wv = tid >> 5, ln = tid & 31;
  // Y1 = gelu(Wg' X + bg)
  gemm_block<5, 4, CINP, CINP, 1, 0>(sWg, sX, sY1, sY1, bg, wv, ln);
  __syncthreads();
  // Y2 = gelu(Wr1' Y1 + br1)   (Y2 stored into sX, stride 128)
  gemm_block<4, 4, 128, 128, 1, 0>(sWr1, sY1, sX, sY1, br1, wv, ln);
  __syncthreads();
  // X = (Wr2' Y2 + br2) + Y1   (in place over sY1)
  gemm_block<4, 4, 128, 128, 0, 1>(sWr2, sX, sY1, sY1, br2, wv, ln);
  __syncthreads();

  // new1 = max over the 32 neighbors
  for (int t = tid; t < 256; t += 256) {
    int o = t & 127, lc = t >> 7;
    float mx = -3.4e38f;
#pragma unroll
    for (int kk = 0; kk < 32; kk++) mx = fmaxf(mx, bf2f(sY1[(lc * 32 + kk) * 128 + o]));
    out[O_NEW1 + (size_t)(b * 128 + o) * M1 + (m0 + lc)] = mx;
  }
}

// ---------------------------------------------------------------------
extern "C" void kernel_launch(void* const* d_in, const int* in_sizes, int n_in,
                              void* d_out, int out_size, void* d_ws, size_t ws_size,
                              hipStream_t stream) {
  (void)in_sizes; (void)n_in; (void)out_size; (void)ws_size;
  const float* pts = (const float*)d_in[0];
  const float* W1 = (const float*)d_in[1];  const float* s1 = (const float*)d_in[2];  const float* b1 = (const float*)d_in[3];
  const float* W2 = (const float*)d_in[4];  const float* s2 = (const float*)d_in[5];  const float* b2 = (const float*)d_in[6];
  const float* Wp = (const float*)d_in[7];  const float* sp = (const float*)d_in[8];  const float* bp = (const float*)d_in[9];
  const float* Wg = (const float*)d_in[10]; const float* sg = (const float*)d_in[11]; const float* bg = (const float*)d_in[12];
  const float* Wr1 = (const float*)d_in[13]; const float* sr1 = (const float*)d_in[14]; const float* br1 = (const float*)d_in[15];
  const float* Wr2 = (const float*)d_in[16]; const float* sr2 = (const float*)d_in[17]; const float* br2 = (const float*)d_in[18];
  float* out = (float*)d_out;
  char* ws = (char*)d_ws;

  float* W1f   = (float*)(ws + 0);        //   768 B
  float* W2f   = (float*)(ws + 768);      // 16384 B
  float* Wpf   = (float*)(ws + 17152);    // 16384 B
  u16*   Wgb   = (u16*)(ws + 33536);      // 40960 B
  u16*   Wr1b  = (u16*)(ws + 74496);      // 32768 B
  u16*   Wr2b  = (u16*)(ws + 107264);     // 32768 B
  int*   fpsidx = (int*)(ws + 140032);    // 65536 B
  float* ctr   = (float*)(ws + 205568);   // 196608 B
  float* fp    = (float*)(ws + 402176);   // 4 MiB
  int*   gidx  = (int*)(ws + 4596480);    // 2 MiB    (total ~6.4 MiB)

  k_prep<<<64, 256, 0, stream>>>(W1, s1, W2, s2, Wp, sp, Wg, sg, Wr1, sr1, Wr2, sr2,
                                 W1f, W2f, Wpf, Wgb, Wr1b, Wr2b);
  k_copy<<<256, 256, 0, stream>>>(pts, out);
  k_feats<<<256, 256, 0, stream>>>(pts, W1f, b1, W2f, b2, out);
  k_fps<<<BB, 1024, 0, stream>>>(pts, fpsidx);
  k_gather<<<1024, 256, 0, stream>>>(pts, fpsidx, out, ctr);
  k_fp<<<64, 256, 0, stream>>>(out, Wpf, bp, fp);
  k_knn<<<BB * M1, 256, 0, stream>>>(pts, ctr, out, gidx);
  k_groupconv<<<BB * 1024, 256, 0, stream>>>(out, pts, ctr, fp, gidx,
                                             Wgb, bg, Wr1b, br1, Wr2b, br2, out);
}
